// TemporalLandmarkAttention_18150531793425
// MI455X (gfx1250) — matrix operand order repaired and checked by reading, not compiled, not run on hardware
//
#include <hip/hip_runtime.h>
#include <hip/hip_bf16.h>

typedef __attribute__((ext_vector_type(16))) _Float16 v16h;
typedef __attribute__((ext_vector_type(8)))  float    v8f;

#define Bn 32
#define Cn 256
#define Tn 256
#define Vn 25
#define KL 8
#define DK 32
#define LSTRIDE (Tn / KL)   // 32

// ---------- WMMA fragment helpers (v_wmma_f32_16x16x32_f16, wave32) ----------
// A (16x32 f16): lane L -> row m0+(L&15); K base (L>>4)*8; elems 0-7 = K kb..kb+7,
// elems 8-15 = K kb+16..kb+23  (per CDNA5 ISA 16-bit A layout table).
__device__ __forceinline__ v16h load_a(const _Float16* __restrict__ src, int ld,
                                       int m0, int k0) {
  int lane = threadIdx.x & 31;
  const _Float16* p = src + (size_t)(m0 + (lane & 15)) * ld + k0 + ((lane >> 4) << 3);
  v16h a;
#pragma unroll
  for (int e = 0; e < 8; ++e) { a[e] = p[e]; a[e + 8] = p[e + 16]; }
  return a;
}

// A fragment where local row r maps to landmark row (r&7)*LSTRIDE of src.
__device__ __forceinline__ v16h load_a_land(const _Float16* __restrict__ src, int ld,
                                            int k0) {
  int lane = threadIdx.x & 31;
  const _Float16* p = src + (size_t)((lane & 7) * LSTRIDE) * ld + k0 + ((lane >> 4) << 3);
  v16h a;
#pragma unroll
  for (int e = 0; e < 8; ++e) { a[e] = p[e]; a[e + 8] = p[e + 16]; }
  return a;
}

// B (32x16 f16): lane L -> col n0+(L&15); K base (L>>4)*8; elems 0-7 = K kb..kb+7, elems 8-15 = K kb+16..kb+23.
// f(k, n) supplies element B[k][n].
template <typename F>
__device__ __forceinline__ v16h load_b(int n0, int k0, F f) {
  int lane = threadIdx.x & 31;
  int n  = n0 + (lane & 15);
  int kb = k0 + ((lane >> 4) << 3);
  v16h b;
#pragma unroll
  for (int e = 0; e < 8; ++e) { b[e] = f(kb + e, n); b[e + 8] = f(kb + 16 + e, n); }
  return b;
}

// D (16x16 f32): VGPR r -> row m0 + r + (lane>>4)*8, col n0 + (lane&15).
__device__ __forceinline__ void store_d_f16(_Float16* __restrict__ dst, int ld,
                                            int m0, int n0, v8f d) {
  int lane = threadIdx.x & 31;
  int n  = n0 + (lane & 15);
  int mb = m0 + ((lane >> 4) << 3);
#pragma unroll
  for (int r = 0; r < 8; ++r) dst[(size_t)(mb + r) * ld + n] = (_Float16)d[r];
}

#define WMMA(acc, a, b) \
  __builtin_amdgcn_wmma_f32_16x16x32_f16(false, (a), false, (b), (short)0, (acc), false, false)

// ---------- Kernel 1: x_t[b,t,c] = mean_v x[b,c,t,v], store f16 ----------
// One block per (b,c): stage 6400-float contiguous row via aligned float4 loads,
// then thread t reduces its 25 LDS floats (25 coprime to 64 banks -> conflict-free).
__global__ __launch_bounds__(256)
void mean_kernel(const float* __restrict__ x, _Float16* __restrict__ xt) {
  __shared__ float row[Tn * Vn];                  // 25.6 KB
  const int bc = blockIdx.x;                      // b*Cn + c
  const int c  = bc % Cn;
  const int b  = bc / Cn;
  const float4* src = (const float4*)(x + (size_t)bc * (Tn * Vn));
  float4* rowv = (float4*)row;
  for (int i = threadIdx.x; i < (Tn * Vn) / 4; i += 256) rowv[i] = src[i];
  __syncthreads();
  const int t = threadIdx.x;
  const float* p = row + t * Vn;
  float s = 0.f;
#pragma unroll
  for (int v = 0; v < Vn; ++v) s += p[v];
  xt[((size_t)b * Tn + t) * Cn + c] = (_Float16)(s * (1.0f / (float)Vn));
}

// ---------- Kernel 2: per-batch landmark attention, all GEMMs via WMMA ----------
__global__ __launch_bounds__(256)
void attn_kernel(const _Float16* __restrict__ xt,
                 const float* __restrict__ Wq, const float* __restrict__ Wk,
                 const float* __restrict__ Wv, const float* __restrict__ Wo,
                 float* __restrict__ outc) {
  __shared__ _Float16 q_lds[Tn * DK];        // 16 KB  q (256x32)
  __shared__ _Float16 k_lds[16 * DK];        //  1 KB  k (16x32, rows 8-15 alias)
  __shared__ _Float16 v_lds[32 * DK];        //  2 KB  v (32x32, rows 16-31 zero)
  __shared__ _Float16 attn_lds[8][16 * DK];  //  8 KB  per-wave attn tile (16x32)
  __shared__ _Float16 out_lds[Tn * DK];      // 16 KB  attn@v (256x32)

  const int tid  = threadIdx.x;
  const int lane = tid & 31;
  const int wave = tid >> 5;
  const int b    = blockIdx.x;
  const _Float16* xtb = xt + (size_t)b * Tn * Cn;

  // zero v rows 16..31 (K-padding for attn@v)
  for (int i = tid; i < 16 * DK; i += 256) v_lds[16 * DK + i] = (_Float16)0.f;

  // ---- Phase 1: q (32 tiles), k (2 tiles), v (2 tiles) ----
  for (int tile = wave; tile < 36; tile += 8) {
    v8f acc = {};
    if (tile < 32) {
      int m0 = (tile >> 1) << 4;
      int n0 = (tile & 1) << 4;
      for (int k0 = 0; k0 < Cn; k0 += 32) {
        v16h a = load_a(xtb, Cn, m0, k0);
        v16h bm = load_b(n0, k0, [&](int k, int n) { return (_Float16)Wq[n * Cn + k]; });
        acc = WMMA(acc, a, bm);
      }
      store_d_f16(q_lds, DK, m0, n0, acc);
    } else {
      const float* W   = (tile < 34) ? Wk : Wv;
      _Float16*    dst = (tile < 34) ? k_lds : v_lds;
      int n0 = (tile & 1) << 4;
      for (int k0 = 0; k0 < Cn; k0 += 32) {
        v16h a = load_a_land(xtb, Cn, k0);
        v16h bm = load_b(n0, k0, [&](int k, int n) { return (_Float16)W[n * Cn + k]; });
        acc = WMMA(acc, a, bm);
      }
      store_d_f16(dst, DK, 0, n0, acc);
    }
  }
  __syncthreads();

  // ---- Phase 2: scores -> softmax(8) -> attn @ v ----
  const float scale = 0.17677669529663687f;  // 1/sqrt(DK)
  for (int mt = wave; mt < 16; mt += 8) {
    int m0 = mt << 4;
    v8f s = {};
    {
      v16h a  = load_a(q_lds, DK, m0, 0);
      v16h bm = load_b(0, 0, [&](int k, int n) { return k_lds[n * DK + k]; });
      s = WMMA(s, a, bm);
    }
    int  j     = lane & 15;
    bool valid = (j < KL);
    _Float16* aw = attn_lds[wave];
#pragma unroll
    for (int r = 0; r < 8; ++r) {
      float sv = s[r] * scale;
      float mx = valid ? sv : -3.0e38f;          // reduce over 8-lane groups (wave32)
      mx = fmaxf(mx, __shfl_xor(mx, 1));
      mx = fmaxf(mx, __shfl_xor(mx, 2));
      mx = fmaxf(mx, __shfl_xor(mx, 4));
      float e  = valid ? __expf(sv - mx) : 0.f;
      float sm = e;
      sm += __shfl_xor(sm, 1);
      sm += __shfl_xor(sm, 2);
      sm += __shfl_xor(sm, 4);
      float av = valid ? (e / sm) : 0.f;
      int ml = r + ((lane >> 4) << 3);
      aw[ml * DK + j]      = (_Float16)av;       // cols 0..15 (8..15 are zero)
      aw[ml * DK + 16 + j] = (_Float16)0.f;      // cols 16..31 zero (K-pad)
    }
#pragma unroll
    for (int nt = 0; nt < 2; ++nt) {
      int n0 = nt << 4;
      v8f o = {};
      v16h a  = load_a(aw, DK, 0, 0);
      v16h bm = load_b(n0, 0, [&](int k, int n) { return v_lds[k * DK + n]; });
      o = WMMA(o, a, bm);
      store_d_f16(out_lds, DK, m0, n0, o);
    }
  }
  __syncthreads();

  // ---- Phase 3: outc = out(256x32) @ Wo^T(32x256) ----
  float* outcb = outc + (size_t)b * Tn * Cn;
  for (int tile = wave; tile < 256; tile += 8) {
    int m0 = (tile >> 4) << 4;
    int n0 = (tile & 15) << 4;
    v8f acc = {};
    v16h a  = load_a(out_lds, DK, m0, 0);
    v16h bm = load_b(n0, 0, [&](int k, int n) { return (_Float16)Wo[n * DK + k]; });
    acc = WMMA(acc, a, bm);
    int n = n0 + (lane & 15);
#pragma unroll
    for (int r = 0; r < 8; ++r) {
      int m = m0 + r + ((lane >> 4) << 3);
      outcb[(size_t)m * Cn + n] = acc[r];
    }
  }
}

// ---------- Kernel 3: out = x + sigmoid(gate) * outc[b,t,c] (broadcast over V) ----------
// One block per (b,c): preload the 256 gated outc values for this column into LDS,
// then stream the contiguous 25.6 KB row with aligned float4 loads/stores.
__global__ __launch_bounds__(256)
void add_kernel(const float* __restrict__ x, const float* __restrict__ outc,
                const float* __restrict__ gate, float* __restrict__ out) {
  __shared__ float oc[Tn];
  const int bc = blockIdx.x;                      // b*Cn + c
  const int c  = bc % Cn;
  const int b  = bc / Cn;
  const float sig = 1.0f / (1.0f + __expf(-gate[0]));
  oc[threadIdx.x] = sig * outc[((size_t)b * Tn + threadIdx.x) * Cn + c];  // L2 hit
  __syncthreads();
  const float4* src = (const float4*)(x   + (size_t)bc * (Tn * Vn));
  float4*       dst = (float4*)      (out + (size_t)bc * (Tn * Vn));
  for (int i = threadIdx.x; i < (Tn * Vn) / 4; i += 256) {
    float4 xv = src[i];
    int e = i * 4;                                // element index within row
    xv.x += oc[(e    ) / Vn];
    xv.y += oc[(e + 1) / Vn];
    xv.z += oc[(e + 2) / Vn];
    xv.w += oc[(e + 3) / Vn];
    dst[i] = xv;
  }
}

extern "C" void kernel_launch(void* const* d_in, const int* in_sizes, int n_in,
                              void* d_out, int out_size, void* d_ws, size_t ws_size,
                              hipStream_t stream) {
  const float* x    = (const float*)d_in[0];
  const float* Wq   = (const float*)d_in[1];
  const float* Wk   = (const float*)d_in[2];
  const float* Wv   = (const float*)d_in[3];
  const float* Wo   = (const float*)d_in[4];
  const float* gate = (const float*)d_in[5];
  float* out = (float*)d_out;

  _Float16* xt   = (_Float16*)d_ws;                                   // 4 MB
  float*    outc = (float*)((char*)d_ws +
                            (size_t)Bn * Tn * Cn * sizeof(_Float16)); // 8 MB

  mean_kernel<<<Bn * Cn, 256, 0, stream>>>(x, xt);
  attn_kernel<<<Bn, 256, 0, stream>>>(xt, Wq, Wk, Wv, Wo, outc);
  add_kernel<<<Bn * Cn, 256, 0, stream>>>(x, outc, gate, out);
}
